// ULR_71992241815552
// MI455X (gfx1250) — matrix operand order repaired and checked by reading, not compiled, not run on hardware
//
#include <hip/hip_runtime.h>
#include <cstdint>

// ---------------------------------------------------------------------------
// ULR fused pipeline for MI455X (gfx1250, wave32, WMMA f32_16x16x32_f16)
//
//   q  = lang_emb[batch]                      [N,D] (gathered on the fly)
//   qw = q @ W                                [N,D]   f16   (b provably cancels)
//   P  = exp(qw @ key_embT)                   [N,V]   f16   (temperature cancels)
//   den[n] = sum_v P[n,v]                     (deterministic 2-stage reduce)
//   out = 0.5*value_emb[batch] + (0.5/den) * (P @ value_emb)
//
// All heavy GEMMs run on v_wmma_f32_16x16x32_f16 with f32 accumulation.
// key/value embeddings are pre-converted to f16 (64 MB each -> L2 resident).
// ---------------------------------------------------------------------------

#define N_TOK 4096
#define VOCAB 32000
#define DIM   1024
#define VBLK  (VOCAB / 128)     // 250 v-blocks in the score kernel
#define DEN_COLS (VBLK * 2)     // 2 column-waves per block -> 500 partials/row

typedef _Float16 f16;
typedef __attribute__((ext_vector_type(16))) _Float16 v16h;
typedef __attribute__((ext_vector_type(8)))  float    v8f;

union Frag { uint32_t u[8]; v16h h; };

__device__ inline uint32_t pack2(float a, float b) {
    union { f16 h[2]; uint32_t u; } r;
    r.h[0] = (f16)a; r.h[1] = (f16)b;
    return r.u;
}
__device__ inline void unpack2(uint32_t v, f16& a, f16& b) {
    union { uint32_t u; f16 h[2]; } r; r.u = v;
    a = r.h[0]; b = r.h[1];
}

// ---------------------------------------------------------------- converts --
__global__ void cvt_f32_f16(const float2* __restrict__ src,
                            uint32_t* __restrict__ dst, int npairs) {
    int i = blockIdx.x * blockDim.x + threadIdx.x;
    int stride = gridDim.x * blockDim.x;
    for (; i < npairs; i += stride) {
        float2 v = src[i];
        dst[i] = pack2(v.x, v.y);
    }
}

// -------------------------------------------------- qw = lang_emb[batch]@W --
// Block tile 64x64, 8 waves (4 along M x 2 along N), each wave 16x32 of C.
__global__ __launch_bounds__(256) void qw_gemm(
    const int* __restrict__ batch, const float* __restrict__ lang_emb,
    const uint32_t* __restrict__ W16, f16* __restrict__ qw) {
    __shared__ uint32_t As[64 * 17];      // [m][k-pair], 17 = 16 pairs + pad
    __shared__ uint32_t Bs[64 * 17];      // Bt[n][k] halves, row = 34 halves
    __shared__ int bidx[64];
    f16* BsT = (f16*)Bs;

    const int tid = threadIdx.x;
    const int m0 = blockIdx.y * 64, n0 = blockIdx.x * 64;
    if (tid < 64) bidx[tid] = batch[m0 + tid];
    __syncthreads();

    const int wid = tid >> 5, lane = tid & 31;
    const int hi = lane >> 4, ln = lane & 15;
    const int wm = wid & 3, wn = wid >> 2;     // rows wm*16, cols wn*32
    const v8f zf = {};
    v8f acc[2] = {zf, zf};

    for (int kt = 0; kt < DIM / 32; ++kt) {
        const int k0 = kt * 32;
        // A: gather 64x32 f32 rows of lang_emb, convert to f16 pairs
        #pragma unroll
        for (int j = 0; j < 4; ++j) {
            int u = tid + 256 * j;
            int row = u >> 4, pc = u & 15;
            const float* s = lang_emb + (size_t)bidx[row] * DIM + k0 + 2 * pc;
            As[row * 17 + pc] = pack2(s[0], s[1]);
        }
        // B: W16 is row-major [D][D] halves; stage transposed -> Bt[n][k]
        #pragma unroll
        for (int j = 0; j < 4; ++j) {
            int u = tid + 256 * j;
            int k = u >> 5, pn = u & 31;                  // pair along n
            uint32_t pr = W16[(size_t)(k0 + k) * (DIM / 2) + (n0 >> 1) + pn];
            f16 h0, h1; unpack2(pr, h0, h1);
            BsT[(2 * pn) * 34 + k]     = h0;
            BsT[(2 * pn + 1) * 34 + k] = h1;
        }
        __syncthreads();

        Frag a;
        #pragma unroll
        for (int i = 0; i < 8; ++i)
            a.u[i] = As[(wm * 16 + ln) * 17 + 8 * (i >> 2) + 4 * hi + (i & 3)];
        #pragma unroll
        for (int ns = 0; ns < 2; ++ns) {
            Frag b;
            #pragma unroll
            for (int i = 0; i < 8; ++i)
                b.u[i] = Bs[(wn * 32 + ns * 16 + ln) * 17 + 8 * (i >> 2) + 4 * hi + (i & 3)];
            acc[ns] = __builtin_amdgcn_wmma_f32_16x16x32_f16(
                false, a.h, false, b.h, (short)0, acc[ns], false, false);
        }
        __syncthreads();
    }
    #pragma unroll
    for (int ns = 0; ns < 2; ++ns)
        #pragma unroll
        for (int j = 0; j < 8; ++j) {
            int mg = m0 + wm * 16 + j + 8 * hi;
            int ng = n0 + wn * 32 + ns * 16 + ln;
            qw[(size_t)mg * DIM + ng] = (f16)acc[ns][j];
        }
}

// -------------------------------------- P = exp(qw @ keyT), partial denoms --
// Block tile 128x128, 8 waves (4 along M x 2 along N), each wave 32x64 of C.
__global__ __launch_bounds__(256) void score_gemm(
    const uint32_t* __restrict__ qw16, const uint32_t* __restrict__ key16,
    f16* __restrict__ P, float* __restrict__ den_part) {
    __shared__ uint32_t As[128 * 17];
    __shared__ uint32_t Bs[128 * 17];     // key rows are already K-contiguous
    const int tid = threadIdx.x;
    const int m0 = blockIdx.y * 128, v0 = blockIdx.x * 128;
    const int wid = tid >> 5, lane = tid & 31;
    const int hi = lane >> 4, ln = lane & 15;
    const int wm = wid >> 1, wn = wid & 1;     // rows wm*32, cols wn*64
    const v8f zf = {};
    v8f acc[2][4] = {{zf, zf, zf, zf}, {zf, zf, zf, zf}};

    for (int kt = 0; kt < DIM / 32; ++kt) {
        const int kp = kt * 16;   // pair offset into a row
        #pragma unroll
        for (int j = 0; j < 8; ++j) {
            int u = tid + 256 * j;
            int row = u >> 4, pc = u & 15;
            As[row * 17 + pc] = qw16[(size_t)(m0 + row) * (DIM / 2) + kp + pc];
            Bs[row * 17 + pc] = key16[(size_t)(v0 + row) * (DIM / 2) + kp + pc];
        }
        __syncthreads();

        Frag a[2];
        #pragma unroll
        for (int ms = 0; ms < 2; ++ms)
            #pragma unroll
            for (int i = 0; i < 8; ++i)
                a[ms].u[i] = As[(wm * 32 + ms * 16 + ln) * 17 +
                                8 * (i >> 2) + 4 * hi + (i & 3)];
        #pragma unroll
        for (int ns = 0; ns < 4; ++ns) {
            Frag b;
            #pragma unroll
            for (int i = 0; i < 8; ++i)
                b.u[i] = Bs[(wn * 64 + ns * 16 + ln) * 17 + 8 * (i >> 2) + 4 * hi + (i & 3)];
            #pragma unroll
            for (int ms = 0; ms < 2; ++ms)
                acc[ms][ns] = __builtin_amdgcn_wmma_f32_16x16x32_f16(
                    false, a[ms].h, false, b.h, (short)0, acc[ms][ns], false, false);
        }
        __syncthreads();
    }

    // epilogue: exp, store P (f16), deterministic per-wave partial row sums
    #pragma unroll
    for (int ms = 0; ms < 2; ++ms) {
        float rs[8];
        #pragma unroll
        for (int j = 0; j < 8; ++j) rs[j] = 0.f;
        #pragma unroll
        for (int ns = 0; ns < 4; ++ns)
            #pragma unroll
            for (int j = 0; j < 8; ++j) {
                float e = __expf(acc[ms][ns][j]);
                int mg = m0 + wm * 32 + ms * 16 + j + 8 * hi;
                int vg = v0 + wn * 64 + ns * 16 + ln;
                P[(size_t)mg * VOCAB + vg] = (f16)e;
                rs[j] += e;
            }
        #pragma unroll
        for (int j = 0; j < 8; ++j) {
            float r = rs[j];                       // 16-lane tree (stays in half)
            r += __shfl_xor(r, 1); r += __shfl_xor(r, 2);
            r += __shfl_xor(r, 4); r += __shfl_xor(r, 8);
            if (ln == 0) {
                int mg = m0 + wm * 32 + ms * 16 + j + 8 * hi;
                den_part[(size_t)mg * DEN_COLS + blockIdx.x * 2 + wn] = r;
            }
        }
    }
}

// ------------------------------------------------- fixed-order den reduce --
__global__ __launch_bounds__(128) void den_reduce(
    const float* __restrict__ den_part, float* __restrict__ den) {
    __shared__ float red[128];
    const int row = blockIdx.x, tid = threadIdx.x;
    float s = 0.f;
    for (int c = tid; c < DEN_COLS; c += 128)
        s += den_part[(size_t)row * DEN_COLS + c];
    red[tid] = s;
    __syncthreads();
    for (int off = 64; off > 0; off >>= 1) {
        if (tid < off) red[tid] += red[tid + off];
        __syncthreads();
    }
    if (tid == 0) den[row] = red[0];
}

// ------------------- out = 0.5*value_emb[batch] + (0.5/den)*(P @ value16) --
// Block tile 64x256 (few column-blocks minimizes re-reads of the 262MB P),
// 8 waves (2 along M x 4 along N), each wave 32x64 of C.
__global__ __launch_bounds__(256) void value_gemm(
    const uint32_t* __restrict__ P16, const uint32_t* __restrict__ val16,
    const float* __restrict__ den, const int* __restrict__ batch,
    const float* __restrict__ value_emb, float* __restrict__ out) {
    __shared__ uint32_t As[64 * 17];
    __shared__ uint32_t Bs[256 * 17];     // Bt[d][v] halves, row = 34 halves
    f16* BsT = (f16*)Bs;
    const int tid = threadIdx.x;
    const int m0 = blockIdx.y * 64, d0 = blockIdx.x * 256;
    const int wid = tid >> 5, lane = tid & 31;
    const int hi = lane >> 4, ln = lane & 15;
    const int wm = wid >> 2, wn = wid & 3;     // rows wm*32, cols wn*64
    const v8f zf = {};
    v8f acc[2][4] = {{zf, zf, zf, zf}, {zf, zf, zf, zf}};

    for (int kt = 0; kt < VOCAB / 32; ++kt) {
        const int k0 = kt * 32;
        #pragma unroll
        for (int j = 0; j < 4; ++j) {
            int u = tid + 256 * j;
            int row = u >> 4, pc = u & 15;
            As[row * 17 + pc] = P16[(size_t)(m0 + row) * (VOCAB / 2) + (k0 >> 1) + pc];
        }
        // value16 is [V][D] halves; stage transposed -> Bt[d][v]
        #pragma unroll
        for (int j = 0; j < 16; ++j) {
            int u = tid + 256 * j;
            int k = u >> 7, pn = u & 127;                 // pair along d
            uint32_t pr = val16[(size_t)(k0 + k) * (DIM / 2) + (d0 >> 1) + pn];
            f16 h0, h1; unpack2(pr, h0, h1);
            BsT[(2 * pn) * 34 + k]     = h0;
            BsT[(2 * pn + 1) * 34 + k] = h1;
        }
        __syncthreads();

        Frag a[2];
        #pragma unroll
        for (int ms = 0; ms < 2; ++ms)
            #pragma unroll
            for (int i = 0; i < 8; ++i)
                a[ms].u[i] = As[(wm * 32 + ms * 16 + ln) * 17 +
                                8 * (i >> 2) + 4 * hi + (i & 3)];
        #pragma unroll
        for (int ns = 0; ns < 4; ++ns) {
            Frag b;
            #pragma unroll
            for (int i = 0; i < 8; ++i)
                b.u[i] = Bs[(wn * 64 + ns * 16 + ln) * 17 + 8 * (i >> 2) + 4 * hi + (i & 3)];
            #pragma unroll
            for (int ms = 0; ms < 2; ++ms)
                acc[ms][ns] = __builtin_amdgcn_wmma_f32_16x16x32_f16(
                    false, a[ms].h, false, b.h, (short)0, acc[ms][ns], false, false);
        }
        __syncthreads();
    }

    // epilogue: divide by den, blend with gathered value_emb
    #pragma unroll
    for (int ms = 0; ms < 2; ++ms)
        #pragma unroll
        for (int j = 0; j < 8; ++j) {
            int mg = m0 + wm * 32 + ms * 16 + j + 8 * hi;
            float inv = 0.5f / den[mg];
            int bi = batch[mg];
            #pragma unroll
            for (int ns = 0; ns < 4; ++ns) {
                int dg = d0 + wn * 64 + ns * 16 + ln;
                out[(size_t)mg * DIM + dg] =
                    0.5f * value_emb[(size_t)bi * DIM + dg] + acc[ms][ns][j] * inv;
            }
        }
}

// ---------------------------------------------------------------------------
extern "C" void kernel_launch(void* const* d_in, const int* in_sizes, int n_in,
                              void* d_out, int out_size, void* d_ws, size_t ws_size,
                              hipStream_t stream) {
    const int*   batch     = (const int*)  d_in[0];
    const float* lang_emb  = (const float*)d_in[1];
    const float* key_emb   = (const float*)d_in[2];
    const float* value_emb = (const float*)d_in[3];
    const float* W         = (const float*)d_in[4];
    // d_in[5] (b): adds q.b per row to every score -> cancels exactly in the
    // exp-normalization; intentionally unused.
    float* out = (float*)d_out;

    // Workspace carve-out (~412 MB total, all fully rewritten every call):
    char* p = (char*)d_ws;
    auto take = [&](size_t bytes) {
        char* q = p; p += (bytes + 255) & ~(size_t)255; return q;
    };
    uint32_t* W16     = (uint32_t*)take((size_t)DIM * DIM * 2);
    uint32_t* key16   = (uint32_t*)take((size_t)VOCAB * DIM * 2);
    uint32_t* val16   = (uint32_t*)take((size_t)VOCAB * DIM * 2);
    uint32_t* qw16    = (uint32_t*)take((size_t)N_TOK * DIM * 2);
    f16*      P       = (f16*)     take((size_t)N_TOK * VOCAB * 2);
    float*    denpart = (float*)   take((size_t)N_TOK * DEN_COLS * 4);
    float*    den     = (float*)   take((size_t)N_TOK * 4);

    cvt_f32_f16<<<1024, 256, 0, stream>>>((const float2*)W, W16, DIM * DIM / 2);
    cvt_f32_f16<<<8192, 256, 0, stream>>>((const float2*)key_emb, key16,
                                          VOCAB * DIM / 2);
    cvt_f32_f16<<<8192, 256, 0, stream>>>((const float2*)value_emb, val16,
                                          VOCAB * DIM / 2);

    qw_gemm<<<dim3(DIM / 64, N_TOK / 64), 256, 0, stream>>>(
        batch, lang_emb, W16, (f16*)qw16);

    score_gemm<<<dim3(VOCAB / 128, N_TOK / 128), 256, 0, stream>>>(
        qw16, key16, P, denpart);

    den_reduce<<<N_TOK, 128, 0, stream>>>(denpart, den);

    value_gemm<<<dim3(DIM / 256, N_TOK / 64), 256, 0, stream>>>(
        (const uint32_t*)P, val16, den, batch, value_emb, out);
}
